// FastMoELayer_62551903699089
// MI455X (gfx1250) — matrix operand order repaired and checked by reading, not compiled, not run on hardware
//
#include <hip/hip_runtime.h>
#include <cstdint>

// ---------------------------------------------------------------------------
// MoE layer for MI455X (gfx1250, wave32, WMMA + async LDS DMA).
// B=4, S=2048, H=1024, E=8, K=2, FF=4096, T=8192, capacity=1024.
// Expert GEMMs: v_wmma_f32_16x16x32_bf16, double-buffered LDS tiles staged
// with global_load_async_to_lds_b128 (ASYNCcnt).
// ---------------------------------------------------------------------------

typedef __attribute__((ext_vector_type(16))) __bf16 bf16x16;
typedef __attribute__((ext_vector_type(8)))  __bf16 bf16x8;
typedef __attribute__((ext_vector_type(8)))  float  floatx8;
typedef unsigned short u16;
typedef unsigned int   u32;
typedef unsigned long long u64;

#define T_TOK 8192
#define HDIM  1024
#define NEXP  8
#define TOPK  2
#define FFDIM 4096
#define CAP   1024

__device__ __forceinline__ u16 f2bf(float f) {
  u32 x = __float_as_uint(f);
  x += 0x7fffu + ((x >> 16) & 1u);   // round-to-nearest-even
  return (u16)(x >> 16);
}

__device__ __forceinline__ float gelu_tanh(float v) {
  float u = 0.7978845608028654f * (v + 0.044715f * v * v * v);
  u = fminf(fmaxf(u, -15.0f), 15.0f);
  float e2u = __expf(2.0f * u);
  float th = (e2u - 1.0f) / (e2u + 1.0f);
  return 0.5f * v * (1.0f + th);
}

// Async global -> LDS copy of 16 bytes per lane (GVS addressing).
__device__ __forceinline__ void async_copy_b128(u32 lds_addr, u64 base, u32 byte_off) {
  asm volatile("global_load_async_to_lds_b128 %0, %1, %2"
               :: "v"(lds_addr), "v"(byte_off), "s"(base)
               : "memory");
}
__device__ __forceinline__ void wait_async0() {
  asm volatile("s_wait_asynccnt 0x0" ::: "memory");
}

// A fragment: 16x32 bf16 (MxK). LDS tile row-major, stride 40 halves.
// Lane l: row = l&15, half g = l>>4. halves 0..7 -> K = g*8+j,
// halves 8..15 -> K = 16+g*8+j (per CDNA5 ISA 16-bit A layout).
__device__ __forceinline__ bf16x16 frag_a(const u16* tile, int lane) {
  int row = lane & 15, g = (lane >> 4) & 1;
  const u16* p = tile + row * 40;
  union { bf16x16 v; bf16x8 h[2]; } u;
  u.h[0] = *reinterpret_cast<const bf16x8*>(p + g * 8);
  u.h[1] = *reinterpret_cast<const bf16x8*>(p + 16 + g * 8);
  return u.v;
}

// B fragment: 32x16 bf16 (KxN). LDS tile stored [n][k], stride 40.
// Lane l: col = l&15, half g = l>>4. halves 0..7 -> K = g*8+j, halves 8..15 -> K = 16+g*8+j.
__device__ __forceinline__ bf16x16 frag_b(const u16* tile, int lane) {
  int col = lane & 15, g = (lane >> 4) & 1;
  const u16* p = tile + col * 40;
  union { bf16x16 v; bf16x8 h[2]; } u;
  u.h[0] = *reinterpret_cast<const bf16x8*>(p + g * 8);
  u.h[1] = *reinterpret_cast<const bf16x8*>(p + 16 + g * 8);
  return u.v;
}

// ---------------------------------------------------------------------------
// Transpose+convert: src [E][R][C] f32  ->  dst [E][C][R] bf16.
// 32x32 tiles via LDS; coalesced on both sides.
// ---------------------------------------------------------------------------
__global__ __launch_bounds__(256) void moe_transpose_bf16_kernel(
    const float* __restrict__ src, u16* __restrict__ dst, int R, int C) {
  __shared__ u16 tile[32][33];
  const float* s = src + (size_t)blockIdx.z * R * C;
  u16* d = dst + (size_t)blockIdx.z * R * C;
  const int r0 = blockIdx.y * 32, c0 = blockIdx.x * 32;
  const int tr = threadIdx.x >> 5;        // 0..7
  const int tc = threadIdx.x & 31;        // 0..31
#pragma unroll
  for (int i = 0; i < 4; ++i) {
    int r = tr + i * 8;
    tile[r][tc] = f2bf(s[(size_t)(r0 + r) * C + c0 + tc]);
  }
  __syncthreads();
#pragma unroll
  for (int i = 0; i < 4; ++i) {
    int c = tr + i * 8;
    d[(size_t)(c0 + c) * R + r0 + tc] = tile[tc][c];
  }
}

// ---------------------------------------------------------------------------
// Tiled grouped GEMM: C[e] = act(A[e] @ B[e]^T + bias[e])
// A:  bf16 [E, M, K] row-major.
// Bt: bf16 [E, N, K] (pre-transposed weights).
// GELU_BF16OUT=1 -> gelu + bf16 out; 0 -> f32 out.
// Block = 256 threads (8 wave32s); tile 128x128; K step 32; double-buffered
// LDS staged with global_load_async_to_lds_b128.
// ---------------------------------------------------------------------------
template <int GELU_BF16OUT>
__global__ __launch_bounds__(256) void moe_gemm_kernel(
    const u16* __restrict__ Aall, const u16* __restrict__ Btall,
    const float* __restrict__ biasAll, void* __restrict__ Call,
    int M, int N, int K) {
  __shared__ u16 As[2 * 128 * 40];
  __shared__ u16 Bs[2 * 128 * 40];

  const int e  = blockIdx.z;
  const u64 Abase = (u64)(uintptr_t)(Aall  + (size_t)e * M * K);
  const u64 Bbase = (u64)(uintptr_t)(Btall + (size_t)e * N * K);
  const float* bias = biasAll + (size_t)e * N;
  const int m0 = blockIdx.y * 128, n0 = blockIdx.x * 128;
  const int lane = threadIdx.x & 31, wave = threadIdx.x >> 5;
  const int wm = (wave >> 1) * 32, wn = (wave & 1) * 64;

  // Issue async copies of one 128x32 A tile + 128x32 B tile into buffer buf.
  auto issue_tile = [&](int k0, int buf) {
    u16* As_b = As + buf * (128 * 40);
    u16* Bs_b = Bs + buf * (128 * 40);
    for (int ci = threadIdx.x; ci < 512; ci += 256) {
      int r = ci >> 2, q = ci & 3;
      u32 lds_a = (u32)(uintptr_t)(As_b + r * 40 + q * 8);
      u32 off_a = (u32)(((size_t)(m0 + r) * K + k0 + q * 8) * sizeof(u16));
      async_copy_b128(lds_a, Abase, off_a);
      u32 lds_b = (u32)(uintptr_t)(Bs_b + r * 40 + q * 8);
      u32 off_b = (u32)(((size_t)(n0 + r) * K + k0 + q * 8) * sizeof(u16));
      async_copy_b128(lds_b, Bbase, off_b);
    }
  };

  floatx8 acc[2][4];
#pragma unroll
  for (int i = 0; i < 2; ++i)
#pragma unroll
    for (int j = 0; j < 4; ++j)
#pragma unroll
      for (int v = 0; v < 8; ++v) acc[i][j][v] = 0.0f;

  const int NT = K / 32;
  issue_tile(0, 0);
  for (int it = 0; it < NT; ++it) {
    const int cur = it & 1;
    wait_async0();       // this wave's tile-it (and earlier) DMA complete
    __syncthreads();     // all waves' DMA complete; prior reads of other buf done
    if (it + 1 < NT) issue_tile((it + 1) * 32, 1 - cur);  // overlap with compute

    const u16* As_b = As + cur * (128 * 40);
    const u16* Bs_b = Bs + cur * (128 * 40);
    bf16x16 a0 = frag_a(As_b + (wm + 0)  * 40, lane);
    bf16x16 a1 = frag_a(As_b + (wm + 16) * 40, lane);
    bf16x16 b0 = frag_b(Bs_b + (wn + 0)  * 40, lane);
    bf16x16 b1 = frag_b(Bs_b + (wn + 16) * 40, lane);
    bf16x16 b2 = frag_b(Bs_b + (wn + 32) * 40, lane);
    bf16x16 b3 = frag_b(Bs_b + (wn + 48) * 40, lane);

    acc[0][0] = __builtin_amdgcn_wmma_f32_16x16x32_bf16(false, a0, false, b0, (short)0, acc[0][0], false, false);
    acc[0][1] = __builtin_amdgcn_wmma_f32_16x16x32_bf16(false, a0, false, b1, (short)0, acc[0][1], false, false);
    acc[0][2] = __builtin_amdgcn_wmma_f32_16x16x32_bf16(false, a0, false, b2, (short)0, acc[0][2], false, false);
    acc[0][3] = __builtin_amdgcn_wmma_f32_16x16x32_bf16(false, a0, false, b3, (short)0, acc[0][3], false, false);
    acc[1][0] = __builtin_amdgcn_wmma_f32_16x16x32_bf16(false, a1, false, b0, (short)0, acc[1][0], false, false);
    acc[1][1] = __builtin_amdgcn_wmma_f32_16x16x32_bf16(false, a1, false, b1, (short)0, acc[1][1], false, false);
    acc[1][2] = __builtin_amdgcn_wmma_f32_16x16x32_bf16(false, a1, false, b2, (short)0, acc[1][2], false, false);
    acc[1][3] = __builtin_amdgcn_wmma_f32_16x16x32_bf16(false, a1, false, b3, (short)0, acc[1][3], false, false);
    __syncthreads();     // compute reads done before next overwrite of buf cur
  }

  // Epilogue. C/D layout: lane l -> col = l&15, rows (l>>4)*8 + v (v = VGPR idx).
  const int col = lane & 15, rg = (lane >> 4) * 8;
#pragma unroll
  for (int i = 0; i < 2; ++i) {
#pragma unroll
    for (int j = 0; j < 4; ++j) {
      int n = n0 + wn + j * 16 + col;
      float bv = bias[n];
#pragma unroll
      for (int v = 0; v < 8; ++v) {
        int m = m0 + wm + i * 16 + rg + v;
        float val = acc[i][j][v] + bv;
        size_t idx = (size_t)e * M * N + (size_t)m * N + n;
        if (GELU_BF16OUT) {
          reinterpret_cast<u16*>(Call)[idx] = f2bf(gelu_tanh(val));
        } else {
          reinterpret_cast<float*>(Call)[idx] = val;
        }
      }
    }
  }
}

// ---------------------------------------------------------------------------
// Init: row_token = -1, z_accum = 0.
// ---------------------------------------------------------------------------
__global__ void moe_init_kernel(int* __restrict__ row_token, float* __restrict__ z_accum) {
  int i = blockIdx.x * blockDim.x + threadIdx.x;
  if (i < NEXP * CAP) row_token[i] = -1;
  if (i == NEXP * CAP) *z_accum = 0.0f;
}

// ---------------------------------------------------------------------------
// Router: one wave32 per token. logits = x[t] @ router_w, softmax, top-2,
// renormalize; accumulate sum(lse^2) for z-loss.
// ---------------------------------------------------------------------------
__global__ __launch_bounds__(256) void moe_router_kernel(
    const float* __restrict__ x, const float* __restrict__ rw,
    int* __restrict__ top_i, float* __restrict__ top_p,
    float* __restrict__ z_accum) {
  __shared__ float rw_s[HDIM * NEXP];  // 32 KB
  for (int i = threadIdx.x; i < HDIM * NEXP; i += 256) rw_s[i] = rw[i];
  __syncthreads();

  const int wave = threadIdx.x >> 5, lane = threadIdx.x & 31;
  const int t = blockIdx.x * 8 + wave;
  const float* xt = x + (size_t)t * HDIM;

  float acc[NEXP];
#pragma unroll
  for (int e = 0; e < NEXP; ++e) acc[e] = 0.0f;

  for (int h = lane; h < HDIM; h += 32) {
    float xv = xt[h];
    const float* r = rw_s + h * NEXP;
#pragma unroll
    for (int e = 0; e < NEXP; ++e) acc[e] += xv * r[e];
  }
#pragma unroll
  for (int m = 16; m >= 1; m >>= 1) {
#pragma unroll
    for (int e = 0; e < NEXP; ++e) acc[e] += __shfl_xor(acc[e], m, 32);
  }

  if (lane == 0) {
    float mx = acc[0];
#pragma unroll
    for (int e = 1; e < NEXP; ++e) mx = fmaxf(mx, acc[e]);
    float p[NEXP], s = 0.0f;
#pragma unroll
    for (int e = 0; e < NEXP; ++e) { p[e] = expf(acc[e] - mx); s += p[e]; }
    float lse = mx + logf(s);
    atomicAdd(z_accum, lse * lse);
    float inv = 1.0f / s;
    int i0 = 0;
#pragma unroll
    for (int e = 1; e < NEXP; ++e) if (p[e] > p[i0]) i0 = e;
    int i1 = (i0 == 0) ? 1 : 0;
#pragma unroll
    for (int e = 0; e < NEXP; ++e) if (e != i0 && p[e] > p[i1]) i1 = e;
    float p0 = p[i0] * inv, p1 = p[i1] * inv;
    float rs = 1.0f / (p0 + p1);
    top_i[t * 2 + 0] = i0;  top_i[t * 2 + 1] = i1;
    top_p[t * 2 + 0] = p0 * rs;  top_p[t * 2 + 1] = p1 * rs;
  }
}

// ---------------------------------------------------------------------------
// Stable rank sort by descending top-1 prob (ties -> lower token index first).
// rank(t) = |{u: p_u > p_t}| + |{u < t: p_u == p_t}|;  sorted_idx[rank] = t.
// ---------------------------------------------------------------------------
__global__ __launch_bounds__(256) void moe_rank_kernel(
    const float* __restrict__ top_p, int* __restrict__ sorted_idx) {
  __shared__ float keys[256];
  const int t = blockIdx.x * 256 + threadIdx.x;
  const float key = top_p[t * 2];
  int rank = 0;
  for (int tile = 0; tile < T_TOK / 256; ++tile) {
    keys[threadIdx.x] = top_p[(tile * 256 + threadIdx.x) * 2];
    __syncthreads();
    for (int j = 0; j < 256; ++j) {
      int u = tile * 256 + j;
      float ku = keys[j];
      rank += (int)((ku > key) || (ku == key && u < t));
    }
    __syncthreads();
  }
  sorted_idx[rank] = t;
}

// ---------------------------------------------------------------------------
// Capacity assignment. Single block of 512 threads; each owns 32 consecutive
// (rank,k) entries. Per-expert exclusive prefix over thread counts, then pos
// assignment, capacity mask, slot/weight per (token,k), row->token map, and
// both scalar losses.
// ---------------------------------------------------------------------------
__global__ __launch_bounds__(512) void moe_assign_kernel(
    const int* __restrict__ sorted_idx, const int* __restrict__ top_i,
    const float* __restrict__ top_p, int* __restrict__ token_slot,
    float* __restrict__ token_w, int* __restrict__ row_token,
    const float* __restrict__ z_accum, float* __restrict__ out_scalars) {
  __shared__ int cnt[512][NEXP];  // 16 KB
  const int tid = threadIdx.x;
  const int base = tid * 32;

  int local[NEXP];
#pragma unroll
  for (int e = 0; e < NEXP; ++e) local[e] = 0;
  for (int i = 0; i < 32; ++i) {
    int idx = base + i, r = idx >> 1, k = idx & 1;
    int t = sorted_idx[r];
    int e = top_i[t * 2 + k];
    local[e]++;
  }
#pragma unroll
  for (int e = 0; e < NEXP; ++e) cnt[tid][e] = local[e];
  __syncthreads();

  int off[NEXP];
#pragma unroll
  for (int e = 0; e < NEXP; ++e) off[e] = 0;
  for (int u = 0; u < tid; ++u)
#pragma unroll
    for (int e = 0; e < NEXP; ++e) off[e] += cnt[u][e];

  for (int i = 0; i < 32; ++i) {
    int idx = base + i, r = idx >> 1, k = idx & 1;
    int t = sorted_idx[r];
    int e = top_i[t * 2 + k];
    int pos = off[e]++;
    bool keep = pos < CAP;
    int slot = keep ? (e * CAP + pos) : -1;
    token_slot[t * 2 + k] = slot;
    token_w[t * 2 + k] = keep ? top_p[t * 2 + k] : 0.0f;
    if (keep) row_token[slot] = t;
  }

  if (tid == 0) {
    int tot[NEXP];
#pragma unroll
    for (int e = 0; e < NEXP; ++e) tot[e] = 0;
    for (int u = 0; u < 512; ++u)
#pragma unroll
      for (int e = 0; e < NEXP; ++e) tot[e] += cnt[u][e];
    float bl = 0.0f;
#pragma unroll
    for (int e = 0; e < NEXP; ++e) {
      int kept = tot[e] < CAP ? tot[e] : CAP;
      float frac = (float)kept / (float)(T_TOK * TOPK);
      float d = frac - 1.0f / (float)NEXP;
      bl += d * d;
    }
    out_scalars[0] = z_accum[0] / (float)T_TOK;   // z_loss
    out_scalars[1] = bl / (float)NEXP;            // balance_loss
  }
}

// ---------------------------------------------------------------------------
// Gather: expert_in[row] = bf16(x[row_token[row]]) or zeros. 4 halves/thread.
// ---------------------------------------------------------------------------
__global__ __launch_bounds__(256) void moe_gather_kernel(
    const float* __restrict__ x, const int* __restrict__ row_token,
    u16* __restrict__ expert_in) {
  int gid = blockIdx.x * 256 + threadIdx.x;
  int el = gid * 4;
  int row = el >> 10, col = el & (HDIM - 1);
  int t = row_token[row];
  ushort4 o;
  if (t >= 0) {
    float4 xv = *reinterpret_cast<const float4*>(x + (size_t)t * HDIM + col);
    o = make_ushort4(f2bf(xv.x), f2bf(xv.y), f2bf(xv.z), f2bf(xv.w));
  } else {
    o = make_ushort4(0, 0, 0, 0);
  }
  *reinterpret_cast<ushort4*>(expert_in + (size_t)row * HDIM + col) = o;
}

// ---------------------------------------------------------------------------
// Combine: out[t] = w0*y[slot0] + w1*y[slot1]  (slot<0 -> dropped).
// ---------------------------------------------------------------------------
__global__ __launch_bounds__(256) void moe_combine_kernel(
    const float* __restrict__ y, const int* __restrict__ token_slot,
    const float* __restrict__ token_w, float* __restrict__ out) {
  int gid = blockIdx.x * 256 + threadIdx.x;
  int el = gid * 4;
  int t = el >> 10, col = el & (HDIM - 1);
  int s0 = token_slot[t * 2 + 0], s1 = token_slot[t * 2 + 1];
  float w0 = token_w[t * 2 + 0], w1 = token_w[t * 2 + 1];
  float a0 = 0.f, a1 = 0.f, a2 = 0.f, a3 = 0.f;
  if (s0 >= 0) {
    float4 v = *reinterpret_cast<const float4*>(y + (size_t)s0 * HDIM + col);
    a0 += w0 * v.x; a1 += w0 * v.y; a2 += w0 * v.z; a3 += w0 * v.w;
  }
  if (s1 >= 0) {
    float4 v = *reinterpret_cast<const float4*>(y + (size_t)s1 * HDIM + col);
    a0 += w1 * v.x; a1 += w1 * v.y; a2 += w1 * v.z; a3 += w1 * v.w;
  }
  *reinterpret_cast<float4*>(out + (size_t)t * HDIM + col) = make_float4(a0, a1, a2, a3);
}

// ---------------------------------------------------------------------------
// Launch
// ---------------------------------------------------------------------------
extern "C" void kernel_launch(void* const* d_in, const int* in_sizes, int n_in,
                              void* d_out, int out_size, void* d_ws, size_t ws_size,
                              hipStream_t stream) {
  (void)in_sizes; (void)n_in; (void)out_size; (void)ws_size;
  const float* x        = (const float*)d_in[0];  // [T, H]
  const float* router_w = (const float*)d_in[1];  // [H, E]
  const float* up_w     = (const float*)d_in[2];  // [E, H, FF]
  const float* up_b     = (const float*)d_in[3];  // [E, FF]
  const float* down_w   = (const float*)d_in[4];  // [E, FF, H]
  const float* down_b   = (const float*)d_in[5];  // [E, H]
  float* out = (float*)d_out;                     // [T*H] + z_loss + balance_loss

  char* base = (char*)d_ws;
  size_t off = 0;
  auto alloc = [&](size_t nbytes) -> void* {
    off = (off + 255) & ~(size_t)255;
    void* p = base + off;
    off += nbytes;
    return p;
  };
  float* z_accum   = (float*)alloc(sizeof(float));
  int*   top_i     = (int*)  alloc((size_t)T_TOK * 2 * sizeof(int));
  float* top_p     = (float*)alloc((size_t)T_TOK * 2 * sizeof(float));
  int*   sortedidx = (int*)  alloc((size_t)T_TOK * sizeof(int));
  int*   tokenslot = (int*)  alloc((size_t)T_TOK * 2 * sizeof(int));
  float* tokenw    = (float*)alloc((size_t)T_TOK * 2 * sizeof(float));
  int*   rowtoken  = (int*)  alloc((size_t)NEXP * CAP * sizeof(int));
  u16*   expert_in = (u16*)  alloc((size_t)NEXP * CAP * HDIM * sizeof(u16));
  u16*   hid       = (u16*)  alloc((size_t)NEXP * CAP * FFDIM * sizeof(u16));
  float* ybuf      = (float*)alloc((size_t)NEXP * CAP * HDIM * sizeof(float));
  u16*   up_wt     = (u16*)  alloc((size_t)NEXP * HDIM * FFDIM * sizeof(u16));  // [E,FF,H] bf16
  u16*   down_wt   = (u16*)  alloc((size_t)NEXP * FFDIM * HDIM * sizeof(u16));  // [E,H,FF] bf16

  // 0) init scratch
  moe_init_kernel<<<(NEXP * CAP + 256) / 256, 256, 0, stream>>>(rowtoken, z_accum);
  // 0b) pre-transpose + bf16-convert expert weights (B operands become [N][K])
  moe_transpose_bf16_kernel<<<dim3(FFDIM / 32, HDIM / 32, NEXP), 256, 0, stream>>>(
      up_w, up_wt, HDIM, FFDIM);
  moe_transpose_bf16_kernel<<<dim3(HDIM / 32, FFDIM / 32, NEXP), 256, 0, stream>>>(
      down_w, down_wt, FFDIM, HDIM);
  // 1) router
  moe_router_kernel<<<T_TOK / 8, 256, 0, stream>>>(x, router_w, top_i, top_p, z_accum);
  // 2) priority sort
  moe_rank_kernel<<<T_TOK / 256, 256, 0, stream>>>(top_p, sortedidx);
  // 3) capacity assignment + losses
  moe_assign_kernel<<<1, 512, 0, stream>>>(sortedidx, top_i, top_p, tokenslot,
                                           tokenw, rowtoken, z_accum,
                                           out + (size_t)T_TOK * HDIM);
  // 4) dispatch gather (f32 -> bf16)
  moe_gather_kernel<<<(NEXP * CAP * HDIM / 4) / 256, 256, 0, stream>>>(x, rowtoken, expert_in);
  // 5) up projection + gelu (bf16 out)
  moe_gemm_kernel<1><<<dim3(FFDIM / 128, CAP / 128, NEXP), 256, 0, stream>>>(
      expert_in, up_wt, up_b, (void*)hid, CAP, FFDIM, HDIM);
  // 6) down projection (f32 out)
  moe_gemm_kernel<0><<<dim3(HDIM / 128, CAP / 128, NEXP), 256, 0, stream>>>(
      hid, down_wt, down_b, (void*)ybuf, CAP, HDIM, FFDIM);
  // 7) weighted combine
  moe_combine_kernel<<<(T_TOK * HDIM / 4) / 256, 256, 0, stream>>>(ybuf, tokenslot, tokenw, out);
}
